// GATLayer_7584912245088
// MI455X (gfx1250) — matrix operand order repaired and checked by reading, not compiled, not run on hardware
//
#include <hip/hip_runtime.h>
#include <hip/hip_bf16.h>

typedef __attribute__((ext_vector_type(16))) _Float16 v16h;
typedef __attribute__((ext_vector_type(8)))  _Float16 v8h;
typedef __attribute__((ext_vector_type(2)))  _Float16 v2h;
typedef __attribute__((ext_vector_type(8)))  float    v8f;

#define HEADS 4
#define CH    32
#define FDIM  128   // input features == H*C == 128
#define EDIM  16
#define NTILES (FDIM / 16)   // 8 column tiles
#define KCHUNKS (FDIM / 32)  // 4 K-chunks of 32

// ---------------------------------------------------------------------------
// Utility: float atomic max via CAS (handles negatives correctly)
// ---------------------------------------------------------------------------
__device__ __forceinline__ void atomicMaxFloat(float* addr, float value) {
    int* iaddr = (int*)addr;
    int old = __float_as_int(*addr);
    while (__int_as_float(old) < value) {
        int assumed = old;
        old = atomicCAS(iaddr, assumed, __float_as_int(value));
        if (old == assumed) break;
    }
}

__global__ void fill_val_kernel(float* p, float v, int n) {
    int i = blockIdx.x * blockDim.x + threadIdx.x;
    if (i < n) p[i] = v;
}

// ---------------------------------------------------------------------------
// K0a: f32 -> f16 pre-conversion of node features (2 elems/thread, packed cvt)
// ---------------------------------------------------------------------------
__global__ void cvt_f16x2_kernel(const float* __restrict__ in,
                                 _Float16* __restrict__ out, int n2) {
    int i = blockIdx.x * blockDim.x + threadIdx.x;
    if (i >= n2) return;
    float2 v = ((const float2*)in)[i];
    v2h o; o[0] = (_Float16)v.x; o[1] = (_Float16)v.y;
    ((v2h*)out)[i] = o;
}

// ---------------------------------------------------------------------------
// K0b: pack W (f32, KxN row-major) into f16 B-register layout:
//      PB[((nt*4 + kc)*32 + lane)*16 + e] = W[kc*32 + (e<8 ? 8*(lane>=16) + e : 16 + 8*(lane>=16) + e-8)][nt*16 + lane%16]
// so each lane's 16 B halves are one contiguous 32B run. 32 KB total.
// ---------------------------------------------------------------------------
__global__ void pack_W_kernel(const float* __restrict__ W,
                              _Float16* __restrict__ PB) {
    int i = blockIdx.x * blockDim.x + threadIdx.x;
    if (i >= NTILES * KCHUNKS * 32 * 16) return;
    int e  = i & 15;
    int l  = (i >> 4) & 31;
    int kc = (i >> 9) & (KCHUNKS - 1);
    int nt = i >> 11;
    int k = kc * 32 + (e < 8 ? 8 * (l >> 4) + e : 16 + 8 * (l >> 4) + (e - 8));
    int n = nt * 16 + (l & 15);
    PB[i] = (_Float16)W[(size_t)k * FDIM + n];
}

// ---------------------------------------------------------------------------
// K1: X[N,128] = NF[N,128] @ W[128,128] via v_wmma_f32_16x16x32_f16.
// One wave per 16x16 output tile; K=128 in 4 steps of 32.
// A (16x32 f16): lane m = lane&15; lanes<16 hold K {0..7,16..23}, lanes>=16
//                hold K {8..15,24..31} (ISA 7.12.2) -> two b128 loads/chunk.
// B: pre-packed layout -> one contiguous 32B load per lane per chunk.
// C (16x16 f32): VGPR r: lanes<16 -> M=r, lanes>=16 -> M=r+8.
// Store epilogue: uniform full-tile branch -> straight 8-store fast path.
// ---------------------------------------------------------------------------
__global__ void proj_wmma_kernel(const _Float16* __restrict__ NFh,
                                 const _Float16* __restrict__ PB,
                                 float* __restrict__ X, int N) {
    const int nt   = blockIdx.x;               // output column tile
    const int m0   = blockIdx.y * 16;          // output row tile
    const int lane = threadIdx.x;              // 0..31 (wave32)
    const int g    = lane >> 4;
    const int lm   = lane & 15;

    int mrow = m0 + lm; if (mrow >= N) mrow = N - 1;   // clamp (N%16==0 normally)
    const _Float16* __restrict__ arow = NFh + (size_t)mrow * FDIM + (g ? 8 : 0);
    const _Float16* __restrict__ brow = PB + ((size_t)nt * KCHUNKS * 32 + lane) * 16;

    v8f acc = {};
    #pragma unroll
    for (int kc = 0; kc < KCHUNKS; ++kc) {
        v8h alo = *(const v8h*)(arow + kc * 32);
        v8h ahi = *(const v8h*)(arow + kc * 32 + 16);
        v16h a, b;
        #pragma unroll
        for (int e = 0; e < 8; ++e) { a[e] = alo[e]; a[8 + e] = ahi[e]; }
        b = *(const v16h*)(brow + (size_t)kc * 32 * 16);

        acc = __builtin_amdgcn_wmma_f32_16x16x32_f16(
            /*neg_a=*/false, a, /*neg_b=*/false, b,
            /*c_mod=*/(short)0, acc, /*reuse_a=*/false, /*reuse_b=*/false);
    }

    float* __restrict__ xout = X + (size_t)(m0 + 8 * g) * FDIM + nt * 16 + lm;
    if (m0 + 16 <= N) {                       // wave-uniform: full tile in range
        #pragma unroll
        for (int r = 0; r < 8; ++r) xout[(size_t)r * FDIM] = acc[r];
    } else {                                  // ragged tail (not hit for N%16==0)
        #pragma unroll
        for (int r = 0; r < 8; ++r) {
            int m = m0 + r + 8 * g;
            if (m < N) X[(size_t)m * FDIM + nt * 16 + lm] = acc[r];
        }
    }
}

// ---------------------------------------------------------------------------
// K2: a_src[n,h], a_dst[n,h] = <X[n,h,:], att_src/dst[h,:]>
// ---------------------------------------------------------------------------
__global__ void att_nodes_kernel(const float* __restrict__ X,
                                 const float* __restrict__ att_src,
                                 const float* __restrict__ att_dst,
                                 float* __restrict__ a_src,
                                 float* __restrict__ a_dst, int N) {
    int i = blockIdx.x * blockDim.x + threadIdx.x;
    if (i >= N * HEADS) return;
    int n = i >> 2, h = i & 3;
    const float* xr = X + (size_t)n * FDIM + h * CH;
    const float* as = att_src + h * CH;
    const float* ad = att_dst + h * CH;
    float s = 0.f, d = 0.f;
    #pragma unroll
    for (int c = 0; c < CH; ++c) { float v = xr[c]; s += v * as[c]; d += v * ad[c]; }
    a_src[i] = s; a_dst[i] = d;
}

// ---------------------------------------------------------------------------
// K3: in-degree count and edge_attr segment-sum per destination (self-loop mean)
// ---------------------------------------------------------------------------
__global__ void edge_deg_kernel(const int* __restrict__ ei,
                                const float* __restrict__ ea,
                                float* __restrict__ cnt,
                                float* __restrict__ lsum, int E) {
    int e = blockIdx.x * blockDim.x + threadIdx.x;
    if (e >= E) return;
    int dst = ei[E + e];
    atomicAdd(&cnt[dst], 1.0f);
    const float* row = ea + (size_t)e * EDIM;
    float* out = lsum + (size_t)dst * EDIM;
    #pragma unroll
    for (int d = 0; d < EDIM; ++d) atomicAdd(&out[d], row[d]);
}

// ---------------------------------------------------------------------------
// K4: w2[d,h] = sum_c W_edge[d, h*32+c] * att_edge[h,c]   (16x4, one block)
// ---------------------------------------------------------------------------
__global__ void make_w2_kernel(const float* __restrict__ W_edge,
                               const float* __restrict__ att_edge,
                               float* __restrict__ w2) {
    int i = threadIdx.x;
    if (i >= EDIM * HEADS) return;
    int d = i >> 2, h = i & 3;
    float s = 0.f;
    #pragma unroll
    for (int c = 0; c < CH; ++c)
        s += W_edge[(size_t)d * FDIM + h * CH + c] * att_edge[h * CH + c];
    w2[d * HEADS + h] = s;
}

// ---------------------------------------------------------------------------
// K5: alpha = LeakyReLU(a_src[src]+a_dst[dst]+ea.w2, 0.2); segment max (CAS)
// rows [0,E) = real edges, [E,E+N) = self loops with mean edge_attr
// ---------------------------------------------------------------------------
__global__ void alpha_max_kernel(const int* __restrict__ ei,
                                 const float* __restrict__ ea,
                                 const float* __restrict__ cnt,
                                 const float* __restrict__ lsum,
                                 const float* __restrict__ w2,
                                 const float* __restrict__ a_src,
                                 const float* __restrict__ a_dst,
                                 float* __restrict__ exbuf,
                                 float* __restrict__ mbuf, int E, int N) {
    int i = blockIdx.x * blockDim.x + threadIdx.x;
    if (i >= E + N) return;
    int src, dst;
    float eav[EDIM];
    if (i < E) {
        src = ei[i]; dst = ei[E + i];
        const float* r = ea + (size_t)i * EDIM;
        #pragma unroll
        for (int d = 0; d < EDIM; ++d) eav[d] = r[d];
    } else {
        int n = i - E; src = n; dst = n;
        float inv = 1.0f / fmaxf(cnt[n], 1.0f);
        const float* r = lsum + (size_t)n * EDIM;
        #pragma unroll
        for (int d = 0; d < EDIM; ++d) eav[d] = r[d] * inv;
    }
    #pragma unroll
    for (int h = 0; h < HEADS; ++h) {
        float aeh = 0.f;
        #pragma unroll
        for (int d = 0; d < EDIM; ++d) aeh += eav[d] * w2[d * HEADS + h];
        float a = a_src[(size_t)src * HEADS + h] + a_dst[(size_t)dst * HEADS + h] + aeh;
        a = (a > 0.f) ? a : 0.2f * a;
        exbuf[(size_t)i * HEADS + h] = a;
        atomicMaxFloat(&mbuf[(size_t)dst * HEADS + h], a);
    }
}

// ---------------------------------------------------------------------------
// K6: ex = exp(alpha - m[dst]); denom[dst] += ex
// ---------------------------------------------------------------------------
__global__ void exp_sum_kernel(const int* __restrict__ ei,
                               float* __restrict__ exbuf,
                               const float* __restrict__ mbuf,
                               float* __restrict__ denom, int E, int N) {
    int i = blockIdx.x * blockDim.x + threadIdx.x;
    if (i >= E + N) return;
    int dst = (i < E) ? ei[E + i] : (i - E);
    #pragma unroll
    for (int h = 0; h < HEADS; ++h) {
        float v = __expf(exbuf[(size_t)i * HEADS + h] - mbuf[(size_t)dst * HEADS + h]);
        exbuf[(size_t)i * HEADS + h] = v;
        atomicAdd(&denom[(size_t)dst * HEADS + h], v);
    }
}

// ---------------------------------------------------------------------------
// K7: out[dst, h, c] += X[src, h, c] * ex/denom[dst,h]   (one block per edge)
// ---------------------------------------------------------------------------
__global__ void scatter_out_kernel(const int* __restrict__ ei,
                                   const float* __restrict__ X,
                                   const float* __restrict__ exbuf,
                                   const float* __restrict__ denom,
                                   float* __restrict__ out, int E, int N) {
    int i = blockIdx.x;          // edge row
    int c = threadIdx.x;         // 0..127
    int h = c >> 5;
    int src, dst;
    if (i < E) { src = ei[i]; dst = ei[E + i]; } else { src = i - E; dst = src; }
    float attn = exbuf[(size_t)i * HEADS + h] / denom[(size_t)dst * HEADS + h];
    atomicAdd(&out[(size_t)dst * FDIM + c], X[(size_t)src * FDIM + c] * attn);
}

// ---------------------------------------------------------------------------
// K8: out = LeakyReLU(out + bias, 0.01)
// ---------------------------------------------------------------------------
__global__ void finalize_kernel(float* __restrict__ out,
                                const float* __restrict__ bias, int N) {
    int i = blockIdx.x * blockDim.x + threadIdx.x;
    if (i >= N * FDIM) return;
    float v = out[i] + bias[i & (FDIM - 1)];
    out[i] = (v > 0.f) ? v : 0.01f * v;
}

// ---------------------------------------------------------------------------
extern "C" void kernel_launch(void* const* d_in, const int* in_sizes, int n_in,
                              void* d_out, int out_size, void* d_ws, size_t ws_size,
                              hipStream_t stream) {
    const float* nf       = (const float*)d_in[0];
    const int*   ei       = (const int*)  d_in[1];
    const float* ea       = (const float*)d_in[2];
    const float* W        = (const float*)d_in[3];
    const float* W_edge   = (const float*)d_in[4];
    const float* att_src  = (const float*)d_in[5];
    const float* att_dst  = (const float*)d_in[6];
    const float* att_edge = (const float*)d_in[7];
    const float* bias     = (const float*)d_in[8];

    const int N = in_sizes[0] / FDIM;
    const int E = in_sizes[1] / 2;
    const int T = E + N;

    // ---- workspace carve-up (256B aligned) ----
    char* ws = (char*)d_ws;
    auto carve = [&](size_t bytes) {
        char* p = ws;
        ws += (bytes + 255) & ~(size_t)255;
        return p;
    };
    float*    X     = (float*)   carve((size_t)N * FDIM * 4);
    _Float16* NFh   = (_Float16*)carve((size_t)N * FDIM * 2);
    _Float16* PB    = (_Float16*)carve((size_t)NTILES * KCHUNKS * 32 * 16 * 2);
    float*    a_src = (float*)   carve((size_t)N * HEADS * 4);
    float*    a_dst = (float*)   carve((size_t)N * HEADS * 4);
    float*    cnt   = (float*)   carve((size_t)N * 4);
    float*    lsum  = (float*)   carve((size_t)N * EDIM * 4);
    float*    w2    = (float*)   carve((size_t)EDIM * HEADS * 4);
    float*    exbuf = (float*)   carve((size_t)T * HEADS * 4);
    float*    mbuf  = (float*)   carve((size_t)N * HEADS * 4);
    float*    denom = (float*)   carve((size_t)N * HEADS * 4);

    // ---- zero / -inf init (graph-capturable) ----
    hipMemsetAsync(d_out, 0, (size_t)out_size * 4, stream);
    hipMemsetAsync(cnt,   0, (size_t)N * 4, stream);
    hipMemsetAsync(lsum,  0, (size_t)N * EDIM * 4, stream);
    hipMemsetAsync(denom, 0, (size_t)N * HEADS * 4, stream);
    {
        int n = N * HEADS;
        fill_val_kernel<<<(n + 255) / 256, 256, 0, stream>>>(
            mbuf, -__builtin_huge_valf(), n);
    }

    // K0: operand staging for WMMA (f16 features + register-layout-packed W)
    {
        int n2 = (N * FDIM) / 2;
        cvt_f16x2_kernel<<<(n2 + 255) / 256, 256, 0, stream>>>(nf, NFh, n2);
        int np = NTILES * KCHUNKS * 32 * 16;
        pack_W_kernel<<<(np + 255) / 256, 256, 0, stream>>>(W, PB);
    }
    // K1: node projection GEMM via WMMA
    {
        dim3 grid(NTILES, (N + 15) / 16);
        proj_wmma_kernel<<<grid, 32, 0, stream>>>(NFh, PB, X, N);
    }
    // K2: per-node attention coefficients
    {
        int n = N * HEADS;
        att_nodes_kernel<<<(n + 255) / 256, 256, 0, stream>>>(
            X, att_src, att_dst, a_src, a_dst, N);
    }
    // K3: degree + edge_attr segment sum
    edge_deg_kernel<<<(E + 255) / 256, 256, 0, stream>>>(ei, ea, cnt, lsum, E);
    // K4: w2 = W_edge . att_edge  (16x4)
    make_w2_kernel<<<1, 64, 0, stream>>>(W_edge, att_edge, w2);
    // K5: alpha + segment max
    alpha_max_kernel<<<(T + 255) / 256, 256, 0, stream>>>(
        ei, ea, cnt, lsum, w2, a_src, a_dst, exbuf, mbuf, E, N);
    // K6: exp + denom
    exp_sum_kernel<<<(T + 255) / 256, 256, 0, stream>>>(ei, exbuf, mbuf, denom, E, N);
    // K7: weighted scatter-aggregate
    scatter_out_kernel<<<T, FDIM, 0, stream>>>(ei, X, exbuf, denom, (float*)d_out, E, N);
    // K8: bias + output LeakyReLU
    {
        int n = N * FDIM;
        finalize_kernel<<<(n + 255) / 256, 256, 0, stream>>>((float*)d_out, bias, N);
    }
}
